// LigGridEncoder_43241730736141
// MI455X (gfx1250) — matrix operand order repaired and checked by reading, not compiled, not run on hardware
//
#include <hip/hip_runtime.h>
#include <hip/hip_bf16.h>

typedef __attribute__((ext_vector_type(16))) _Float16 v16h;
typedef __attribute__((ext_vector_type(8)))  _Float16 v8h;
typedef __attribute__((ext_vector_type(8)))  float    v8f;

#define BATCH     131072
#define CHANNELS  32
#define GRID_N    41
#define ENC_FEATS 128
#define MAX_VAL   5.0f
#define PTS_PER_BLOCK 256

// grid strides (floats): [C][X][Y][Z]
#define STRIDE_C (GRID_N * GRID_N * GRID_N)   // 68921
#define STRIDE_X (GRID_N * GRID_N)            // 1681
#define STRIDE_Y (GRID_N)                     // 41

__device__ __forceinline__ void bspline_axis(float v, float* __restrict__ w,
                                             int* __restrict__ idx) {
    float x = 0.5f + 0.5f * (v * (1.0f / MAX_VAL));
    x = fminf(fmaxf(x, 0.0f), 1.0f);
    float u  = x * (float)(GRID_N - 1);
    float fb = floorf(u);
    int   b  = (int)fb;
    float t  = u - fb;
    float t2 = t * t;
    float t3 = t2 * t;
    const float s = 1.0f / 6.0f;
    w[0] = (1.0f - 3.0f * t + 3.0f * t2 - t3) * s;
    w[1] = (4.0f - 6.0f * t2 + 3.0f * t3) * s;
    w[2] = (1.0f + 3.0f * t + 3.0f * t2 - 3.0f * t3) * s;
    w[3] = t3 * s;
#pragma unroll
    for (int k = 0; k < 4; ++k) {
        int ii = b - 1 + k;
        ii = ii < 0 ? 0 : ii;
        ii = ii > (GRID_N - 1) ? (GRID_N - 1) : ii;
        idx[k] = ii;
    }
}

__global__ __launch_bounds__(PTS_PER_BLOCK)
void lig_grid_encoder_kernel(const float* __restrict__ nb_params,   // [BATCH][3]
                             const float* __restrict__ lig_grid,    // [32][41][41][41]
                             const float* __restrict__ enc_w,       // [128][32]
                             const float* __restrict__ enc_b,       // [128]
                             float* __restrict__ out)               // [BATCH][128]
{
    __shared__ _Float16 smem[PTS_PER_BLOCK * CHANNELS];   // 16 KB: lig_x tile, f16

    const int tid = threadIdx.x;
    const int pt  = blockIdx.x * PTS_PER_BLOCK + tid;     // BATCH % 256 == 0

    // ---------------- Phase 1: cubic B-spline gather (L2-resident grid) -----
    float wx[4], wy[4], wz[4];
    int   ix[4], iy[4], iz[4];
    bspline_axis(nb_params[pt * 3 + 0], wx, ix);
    bspline_axis(nb_params[pt * 3 + 1], wy, iy);
    bspline_axis(nb_params[pt * 3 + 2], wz, iz);

    float wxy[16];
    int   offxy[16];
#pragma unroll
    for (int i = 0; i < 4; ++i)
#pragma unroll
        for (int j = 0; j < 4; ++j) {
            wxy[i * 4 + j]   = wx[i] * wy[j];
            offxy[i * 4 + j] = ix[i] * STRIDE_X + iy[j] * STRIDE_Y;
        }
    const int iz0 = iz[0], iz1 = iz[1], iz2 = iz[2], iz3 = iz[3];
    const float wz0 = wz[0], wz1 = wz[1], wz2 = wz[2], wz3 = wz[3];

    float acc[CHANNELS];
    for (int c = 0; c < CHANNELS; ++c) {
        const float* g = lig_grid + (size_t)c * STRIDE_C;
        float s = 0.0f;
#pragma unroll
        for (int q = 0; q < 16; ++q) {
            const float* p = g + offxy[q];
            float zdot = p[iz0] * wz0 + p[iz1] * wz1 + p[iz2] * wz2 + p[iz3] * wz3;
            s += wxy[q] * zdot;
        }
        acc[c] = s;
    }

    // Store row-major f16 lig_x tile to LDS (4 x ds_store_b128 per thread).
    _Float16* row = &smem[tid * CHANNELS];
#pragma unroll
    for (int i = 0; i < 4; ++i) {
        v8h h;
#pragma unroll
        for (int j = 0; j < 8; ++j) h[j] = (_Float16)acc[i * 8 + j];
        *(v8h*)(row + i * 8) = h;
    }

    __syncthreads();

    // ---------------- Phase 2: encoder GEMM via V_WMMA_F32_16X16X32_F16 -----
    const int lane = tid & 31;
    const int wv   = tid >> 5;        // 8 waves
    const int n    = lane & 15;       // N within 16-wide tile / M within A tile
    const int hi   = lane >> 4;       // lane half select

    // B fragments: B[k][n] = enc_w[n_global][k]; halves 0..15 = K kbase+0..15
    v16h  bf[8];
    float bias[8];
    const int kbase = hi * 16;
#pragma unroll
    for (int t = 0; t < 8; ++t) {
        const float* wp = enc_w + (size_t)(t * 16 + n) * CHANNELS + kbase;
        v16h b;
#pragma unroll
        for (int h = 0; h < 16; ++h) b[h] = (_Float16)wp[h];
        bf[t]   = b;
        bias[t] = enc_b[t * 16 + n];
    }

    // Each wave owns two 16-row M-tiles of this block's 256 points.
#pragma unroll
    for (int mt = wv * 2; mt < wv * 2 + 2; ++mt) {
        // A fragment from LDS: lane holds M = lane%16;
        // halves 0-15 = K (hi?16:0)+0..15
        const _Float16* ap = &smem[(mt * 16 + n) * CHANNELS + hi * 16];
        v8h lo = *(const v8h*)ap;
        v8h hh = *(const v8h*)(ap + 8);
        v16h a;
#pragma unroll
        for (int h = 0; h < 8; ++h) { a[h] = lo[h]; a[h + 8] = hh[h]; }

        const size_t mbase = (size_t)blockIdx.x * PTS_PER_BLOCK + mt * 16 + hi * 8;

#pragma unroll
        for (int t = 0; t < 8; ++t) {
            v8f c;
#pragma unroll
            for (int r = 0; r < 8; ++r) c[r] = bias[t];   // bias depends only on N
            // D = A(16x32 f16) x B(32x16 f16) + C(f32)
            c = __builtin_amdgcn_wmma_f32_16x16x32_f16(
                    /*neg_a=*/false, a, /*neg_b=*/false, bf[t],
                    /*c_mod=*/(short)0, c, /*reuse_a=*/false, /*reuse_b=*/false);

            float* op = out + mbase * ENC_FEATS + t * 16 + n;
#pragma unroll
            for (int r = 0; r < 8; ++r) op[(size_t)r * ENC_FEATS] = c[r];
        }
    }
}

extern "C" void kernel_launch(void* const* d_in, const int* in_sizes, int n_in,
                              void* d_out, int out_size, void* d_ws, size_t ws_size,
                              hipStream_t stream) {
    const float* nb_params = (const float*)d_in[0];  // [BATCH][3]
    const float* lig_grid  = (const float*)d_in[1];  // [32][41][41][41]
    const float* enc_w     = (const float*)d_in[2];  // [128][32]
    const float* enc_b     = (const float*)d_in[3];  // [128]
    float*       out       = (float*)d_out;          // [BATCH][128]

    const int batch  = in_sizes[0] / 3;              // 131072
    const int blocks = batch / PTS_PER_BLOCK;        // 512

    lig_grid_encoder_kernel<<<blocks, PTS_PER_BLOCK, 0, stream>>>(
        nb_params, lig_grid, enc_w, enc_b, out);
}
